// Block_4612794876618
// MI455X (gfx1250) — matrix operand order repaired and checked by reading, not compiled, not run on hardware
//
#include <hip/hip_runtime.h>
#include <hip/hip_bf16.h>
#include <math.h>

// ---------------------------------------------------------------------------
// DiT block for MI455X (gfx1250): all matmuls via v_wmma_f32_16x16x32_bf16,
// attention K/V tiles staged through LDS with GLOBAL_LOAD_ASYNC_TO_LDS_B128
// (ASYNCcnt double buffering). B=8, L=1024, D=384, H=6, hd=64.
// ---------------------------------------------------------------------------

typedef __attribute__((ext_vector_type(16))) __bf16 v16bf;
typedef __attribute__((ext_vector_type(8)))  __bf16 v8bf;
typedef __attribute__((ext_vector_type(8)))  float  v8f;

union ABUnion { v16bf v; v8bf h[2]; };

// A-matrix 16x32 bf16 frag, contiguous k order (same as B frag): lane = 16*hi + m.
// lanes 0-15 hold K = k0..k0+15; lanes 16-31 hold K = k0+16..k0+31.
__device__ __forceinline__ v16bf load_a_frag(const __bf16* base, int ld, int k0) {
  int lane = threadIdx.x & 31;
  int m = lane & 15, hi = lane >> 4;
  const __bf16* p = base + (size_t)m * ld + k0 + 16 * hi;
  ABUnion u;
  u.h[0] = *(const v8bf*)(p);
  u.h[1] = *(const v8bf*)(p + 8);
  return u.v;
}

// B-matrix 32x16 bf16 frag: lane n = lane&15; lanes 0-15 hold K=k0..k0+15,
// lanes 16-31 hold K=k0+16..k0+31 (contiguous per lane -> ideal for [N,K] weights).
__device__ __forceinline__ v16bf load_b_frag(const __bf16* base, int ld, int k0) {
  int lane = threadIdx.x & 31;
  int n = lane & 15, hi = lane >> 4;
  const __bf16* p = base + (size_t)n * ld + k0 + 16 * hi;
  ABUnion u;
  u.h[0] = *(const v8bf*)(p);
  u.h[1] = *(const v8bf*)(p + 8);
  return u.v;
}

__device__ __forceinline__ v8f wmma_bf16(v16bf a, v16bf b, v8f c) {
  return __builtin_amdgcn_wmma_f32_16x16x32_bf16(false, a, false, b, (short)0, c,
                                                 false, false);
}

// gfx1250 async copy global->LDS, 16B per lane, GV mode (ASYNCcnt tracked).
__device__ __forceinline__ void async_copy_b128(unsigned lds_off, const __bf16* g) {
  asm volatile("global_load_async_to_lds_b128 %0, %1, off"
               :: "v"(lds_off), "v"((unsigned long long)(size_t)g)
               : "memory");
}
__device__ __forceinline__ void wait_asynccnt0() {
  asm volatile("s_wait_asynccnt 0x0" ::: "memory");
}
__device__ __forceinline__ unsigned lds_offset_of(const void* p) {
  return (unsigned)(size_t)p;   // generic LDS ptr: addr[31:0] is the LDS offset
}

// ---------------------------------------------------------------------------
// Small helper kernels
// ---------------------------------------------------------------------------
__global__ void cvt_bf16_kernel(const float* __restrict__ src,
                                __bf16* __restrict__ dst, int n) {
  int i = blockIdx.x * blockDim.x + threadIdx.x;
  if (i < n) dst[i] = (__bf16)src[i];
}

__global__ void silu_kernel(const float* __restrict__ src,
                            float* __restrict__ dst, int n) {
  int i = blockIdx.x * blockDim.x + threadIdx.x;
  if (i < n) {
    float v = src[i];
    dst[i] = v / (1.0f + __expf(-v));
  }
}

// c[b][n] = silu(cond)[b] . cond_w[n] + cond_b[n]   (8 x 2304, tiny)
__global__ void cond_gemm_kernel(const float* __restrict__ sc,
                                 const float* __restrict__ w,
                                 const float* __restrict__ bias,
                                 float* __restrict__ c) {
  int tid = blockIdx.x * blockDim.x + threadIdx.x;   // 8*2304 threads
  int b = tid / 2304, n = tid % 2304;
  const float* a  = sc + b * 384;
  const float* wr = w  + (size_t)n * 384;
  float acc = bias[n];
  for (int k = 0; k < 384; ++k) acc = fmaf(a[k], wr[k], acc);
  c[tid] = acc;
}

// LayerNorm(last dim 384) + modulate -> bf16. One wave32 per row.
__global__ __launch_bounds__(256) void ln_mod_kernel(
    const float* __restrict__ x, const float* __restrict__ c,
    int s_off, int b_off, __bf16* __restrict__ h) {
  int wave = threadIdx.x >> 5, lane = threadIdx.x & 31;
  int row = blockIdx.x * 8 + wave;                   // 8192 rows total
  const float* xr = x + (size_t)row * 384;
  float v[12], sum = 0.f, sq = 0.f;
#pragma unroll
  for (int i = 0; i < 12; ++i) {
    v[i] = xr[lane + 32 * i];
    sum += v[i];
    sq  += v[i] * v[i];
  }
#pragma unroll
  for (int off = 16; off >= 1; off >>= 1) {
    sum += __shfl_xor(sum, off);
    sq  += __shfl_xor(sq,  off);
  }
  float mu  = sum * (1.0f / 384.0f);
  float var = sq * (1.0f / 384.0f) - mu * mu;        // biased var (matches jnp.var)
  float rs  = rsqrtf(var + 1e-5f);
  int batch = row >> 10;
  const float* sp = c + batch * 2304 + s_off;
  const float* bp = c + batch * 2304 + b_off;
  __bf16* hr = h + (size_t)row * 384;
#pragma unroll
  for (int i = 0; i < 12; ++i) {
    int n = lane + 32 * i;
    float t = (v[i] - mu) * rs;
    hr[n] = (__bf16)(t * (1.0f + sp[n]) + bp[n]);
  }
}

// vt[b][h][d][j] = v[b][j][h*64+d]  (pre-transpose V so PV B-frags are contiguous)
__global__ void transpose_v_kernel(const __bf16* __restrict__ qkv,
                                   __bf16* __restrict__ vt) {
  int tid = blockIdx.x * blockDim.x + threadIdx.x;   // 8*6*64*1024
  int j = tid & 1023;
  int d = (tid >> 10) & 63;
  int rest = tid >> 16;                               // b*6 + h
  int hh = rest % 6, b = rest / 6;
  vt[((size_t)rest * 64 + d) * 1024 + j] =
      qkv[((size_t)b * 1024 + j) * 1152 + 768 + hh * 64 + d];
}

// ---------------------------------------------------------------------------
// bf16 WMMA GEMM: out = A[M,K] * W[N,K]^T + bias, fused epilogue.
// mode 0: store bf16;  mode 1: SiLU -> bf16;
// mode 2: out_f32 = xin + gate[batch][n] * val   (gated residual, N==384)
// Block = 128 thr = 4 waves; wave -> 16x32 tile (A reused over 2 B-frags,
// two independent WMMA chains); k-loop software-pipelined so the compiler
// can overlap next-tile loads with the current WMMAs instead of waitcnt 0.
// ---------------------------------------------------------------------------
__global__ __launch_bounds__(128) void gemm_bf16_kernel(
    const __bf16* __restrict__ A, const __bf16* __restrict__ W,
    const float* __restrict__ bias, int M, int N, int K, int mode,
    __bf16* __restrict__ out_bf, float* __restrict__ out_f,
    const float* __restrict__ xin, const float* __restrict__ gate) {
  int wave = threadIdx.x >> 5;
  int lane = threadIdx.x & 31;
  int laneN = lane & 15, hi = lane >> 4;
  int m0 = blockIdx.x * 64 + wave * 16;
  int n0 = blockIdx.y * 32;
  const __bf16* arow = A + (size_t)m0 * K;
  const __bf16* w0 = W + (size_t)n0 * K;
  const __bf16* w1 = W + (size_t)(n0 + 16) * K;
  v8f acc0 = {}, acc1 = {};
  v16bf a  = load_a_frag(arow, K, 0);
  v16bf b0 = load_b_frag(w0, K, 0);
  v16bf b1 = load_b_frag(w1, K, 0);
  for (int k0 = 32; k0 < K; k0 += 32) {
    v16bf an  = load_a_frag(arow, K, k0);
    v16bf bn0 = load_b_frag(w0, K, k0);
    v16bf bn1 = load_b_frag(w1, K, k0);
    acc0 = wmma_bf16(a, b0, acc0);
    acc1 = wmma_bf16(a, b1, acc1);
    a = an; b0 = bn0; b1 = bn1;
  }
  acc0 = wmma_bf16(a, b0, acc0);
  acc1 = wmma_bf16(a, b1, acc1);

  int na = n0 + laneN, nb = n0 + 16 + laneN;
  float bva = bias[na], bvb = bias[nb];
  if (mode == 0) {
#pragma unroll
    for (int r = 0; r < 8; ++r) {
      int m = m0 + r + 8 * hi;
      out_bf[(size_t)m * N + na] = (__bf16)(acc0[r] + bva);
      out_bf[(size_t)m * N + nb] = (__bf16)(acc1[r] + bvb);
    }
  } else if (mode == 1) {
#pragma unroll
    for (int r = 0; r < 8; ++r) {
      int m = m0 + r + 8 * hi;
      float va = acc0[r] + bva, vb = acc1[r] + bvb;
      out_bf[(size_t)m * N + na] = (__bf16)(va / (1.0f + __expf(-va)));
      out_bf[(size_t)m * N + nb] = (__bf16)(vb / (1.0f + __expf(-vb)));
    }
  } else {
#pragma unroll
    for (int r = 0; r < 8; ++r) {
      int m = m0 + r + 8 * hi;
      int batch = m >> 10;
      float va = acc0[r] + bva, vb = acc1[r] + bvb;
      out_f[(size_t)m * N + na] =
          xin[(size_t)m * N + na] + gate[batch * 2304 + na] * va;
      out_f[(size_t)m * N + nb] =
          xin[(size_t)m * N + nb] + gate[batch * 2304 + nb] * vb;
    }
  }
}

// ---------------------------------------------------------------------------
// Flash attention: grid (L/64, H, B), 4 waves; wave = 16 queries x all keys.
// K/V tiles (shared by all 4 waves) are double-buffer staged into LDS with
// GLOBAL_LOAD_ASYNC_TO_LDS_B128 (+ s_wait_asynccnt + barrier); QK^T and PV on
// WMMA bf16 from ds_load_b128 frags; online softmax fp32; P transposed to
// A-layout through a 16B-aligned LDS tile (same-wave LDS ops are in-order).
// ---------------------------------------------------------------------------
__global__ __launch_bounds__(128) void attn_kernel(
    const __bf16* __restrict__ qkv, const __bf16* __restrict__ vt,
    __bf16* __restrict__ attn_out) {
  constexpr int LDQ = 1152, L = 1024, D = 384;
  __shared__ __align__(16) __bf16 k_lds[2][32][72];  // [buf][key][dim], 144B rows
  __shared__ __align__(16) __bf16 v_lds[2][64][40];  // [buf][dim][key],  80B rows
  __shared__ __align__(16) __bf16 plds[4][16][48];   // per-wave P transpose tile

  int wave = threadIdx.x >> 5, lane = threadIdx.x & 31;
  int laneN = lane & 15, hi = lane >> 4;
  int h = blockIdx.y, b = blockIdx.z;
  int q0 = blockIdx.x * 64 + wave * 16;
  const __bf16* qbase = qkv + ((size_t)b * L + q0) * LDQ + h * 64;
  const __bf16* kbase = qkv + (size_t)b * L * LDQ + 384 + h * 64;
  const __bf16* vbase = vt + (size_t)(b * 6 + h) * 64 * L;   // vbase[d*L + j]

  // stage one 32-key K tile + V tile into LDS buffer `buf` (4 b128/lane)
  auto stage = [&](int buf, int j0) {
    int t = threadIdx.x;
#pragma unroll
    for (int r = 0; r < 2; ++r) {                 // K: 256 x 16B segments
      int e = t * 2 + r;
      int key = e >> 3, seg = e & 7;
      async_copy_b128(lds_offset_of(&k_lds[buf][key][seg * 8]),
                      kbase + (size_t)(j0 + key) * LDQ + seg * 8);
    }
#pragma unroll
    for (int r = 0; r < 2; ++r) {                 // V: 256 x 16B segments
      int e = t * 2 + r;
      int d = e >> 2, seg = e & 3;
      async_copy_b128(lds_offset_of(&v_lds[buf][d][seg * 8]),
                      vbase + (size_t)d * L + j0 + seg * 8);
    }
  };
  // B-frag from LDS K tile: n = key (keyoff+laneN), K-dim = head dim
  auto kfrag = [&](int buf, int keyoff, int k0) -> v16bf {
    ABUnion u;
    const __bf16* p = &k_lds[buf][keyoff + laneN][k0 + 16 * hi];
    u.h[0] = *(const v8bf*)(p);
    u.h[1] = *(const v8bf*)(p + 8);
    return u.v;
  };
  // B-frag from LDS V tile: n = out dim (d0+laneN), K-dim = 32 local keys
  auto vfrag = [&](int buf, int d0) -> v16bf {
    ABUnion u;
    const __bf16* p = &v_lds[buf][d0 + laneN][16 * hi];
    u.h[0] = *(const v8bf*)(p);
    u.h[1] = *(const v8bf*)(p + 8);
    return u.v;
  };

  v16bf aq0 = load_a_frag(qbase, LDQ, 0);
  v16bf aq1 = load_a_frag(qbase, LDQ, 32);

  float mi[8], li[8];
  v8f acc0 = {}, acc1 = {}, acc2 = {}, acc3 = {};
#pragma unroll
  for (int r = 0; r < 8; ++r) { mi[r] = -1e30f; li[r] = 0.f; }
  const float scale = 0.125f;   // 64^-0.5

  stage(0, 0);
  for (int it = 0; it < L / 32; ++it) {
    int cur = it & 1;
    wait_asynccnt0();           // our async stores for buf `cur` have landed
    __syncthreads();            // everyone's have landed
    if (it + 1 < L / 32) stage(cur ^ 1, (it + 1) * 32);  // overlap next tile

    // ---- scores: S = scale * Q K^T   (two 16-key tiles)
    v8f s0 = {}, s1 = {};
    s0 = wmma_bf16(aq0, kfrag(cur, 0, 0),   s0);
    s0 = wmma_bf16(aq1, kfrag(cur, 0, 32),  s0);
    s1 = wmma_bf16(aq0, kfrag(cur, 16, 0),  s1);
    s1 = wmma_bf16(aq1, kfrag(cur, 16, 32), s1);

    float p0[8], p1[8], tmax[8];
#pragma unroll
    for (int r = 0; r < 8; ++r) {
      p0[r] = s0[r] * scale;
      p1[r] = s1[r] * scale;
      tmax[r] = fmaxf(p0[r], p1[r]);
    }
    // row-max across the 16 lanes holding each row (xor 1,2,4,8 stays in half)
#pragma unroll
    for (int off = 1; off < 16; off <<= 1)
#pragma unroll
      for (int r = 0; r < 8; ++r) tmax[r] = fmaxf(tmax[r], __shfl_xor(tmax[r], off));

    float corr[8], rs[8];
#pragma unroll
    for (int r = 0; r < 8; ++r) {
      float nm = fmaxf(mi[r], tmax[r]);
      corr[r] = __expf(mi[r] - nm);
      mi[r] = nm;
      p0[r] = __expf(p0[r] - nm);
      p1[r] = __expf(p1[r] - nm);
      rs[r] = p0[r] + p1[r];
    }
#pragma unroll
    for (int off = 1; off < 16; off <<= 1)
#pragma unroll
      for (int r = 0; r < 8; ++r) rs[r] += __shfl_xor(rs[r], off);
#pragma unroll
    for (int r = 0; r < 8; ++r) {
      li[r] = li[r] * corr[r] + rs[r];
      acc0[r] *= corr[r]; acc1[r] *= corr[r];
      acc2[r] *= corr[r]; acc3[r] *= corr[r];
    }

    // ---- P (C-layout fp32) -> LDS -> A-layout bf16 frag (wave-private tile)
#pragma unroll
    for (int r = 0; r < 8; ++r) {
      plds[wave][r + 8 * hi][laneN]      = (__bf16)p0[r];
      plds[wave][r + 8 * hi][16 + laneN] = (__bf16)p1[r];
    }
    asm volatile("" ::: "memory");     // keep LDS reads after writes (HW in-order)
    ABUnion u;
    const __bf16* pp = &plds[wave][laneN][16 * hi];
    u.h[0] = *(const v8bf*)(pp);
    u.h[1] = *(const v8bf*)(pp + 8);
    v16bf pa = u.v;
    asm volatile("" ::: "memory");

    // ---- O += P V
    acc0 = wmma_bf16(pa, vfrag(cur, 0),  acc0);
    acc1 = wmma_bf16(pa, vfrag(cur, 16), acc1);
    acc2 = wmma_bf16(pa, vfrag(cur, 32), acc2);
    acc3 = wmma_bf16(pa, vfrag(cur, 48), acc3);

    __syncthreads();            // all waves done reading buf `cur` before reuse
  }

  __bf16* obase = attn_out + ((size_t)b * L + q0) * D + h * 64;
#pragma unroll
  for (int r = 0; r < 8; ++r) {
    float inv = 1.0f / li[r];
    int m = r + 8 * hi;
    obase[(size_t)m * D + laneN]      = (__bf16)(acc0[r] * inv);
    obase[(size_t)m * D + 16 + laneN] = (__bf16)(acc1[r] * inv);
    obase[(size_t)m * D + 32 + laneN] = (__bf16)(acc2[r] * inv);
    obase[(size_t)m * D + 48 + laneN] = (__bf16)(acc3[r] * inv);
  }
}

// ---------------------------------------------------------------------------
// Host orchestration
// ---------------------------------------------------------------------------
extern "C" void kernel_launch(void* const* d_in, const int* in_sizes, int n_in,
                              void* d_out, int out_size, void* d_ws, size_t ws_size,
                              hipStream_t stream) {
  (void)in_sizes; (void)n_in; (void)out_size; (void)ws_size;
  const int B = 8, L = 1024, D = 384, M = B * L;

  const float* x      = (const float*)d_in[0];
  const float* cond   = (const float*)d_in[1];
  const float* qkv_w  = (const float*)d_in[2];
  const float* qkv_b  = (const float*)d_in[3];
  const float* proj_w = (const float*)d_in[4];
  const float* proj_b = (const float*)d_in[5];
  const float* mlp_w1 = (const float*)d_in[6];
  const float* mlp_b1 = (const float*)d_in[7];
  const float* mlp_w2 = (const float*)d_in[8];
  const float* mlp_b2 = (const float*)d_in[9];
  const float* cond_w = (const float*)d_in[10];
  const float* cond_b = (const float*)d_in[11];
  float* out = (float*)d_out;

  // workspace carve-up (256B aligned)
  char* ws = (char*)d_ws;
  size_t off = 0;
  auto carve = [&](size_t bytes) -> char* {
    char* p = ws + off;
    off += (bytes + 255) & ~(size_t)255;
    return p;
  };
  float*  sc    = (float*)carve(B * D * 4);
  float*  cbuf  = (float*)carve(B * 6 * D * 4);
  __bf16* wqkv  = (__bf16*)carve((size_t)3 * D * D * 2);
  __bf16* wproj = (__bf16*)carve((size_t)D * D * 2);
  __bf16* wm1   = (__bf16*)carve((size_t)4 * D * D * 2);
  __bf16* wm2   = (__bf16*)carve((size_t)4 * D * D * 2);
  __bf16* h1    = (__bf16*)carve((size_t)M * D * 2);
  __bf16* qkvb  = (__bf16*)carve((size_t)M * 3 * D * 2);
  __bf16* vtb   = (__bf16*)carve((size_t)M * D * 2);
  __bf16* attnb = (__bf16*)carve((size_t)M * D * 2);
  float*  x2    = (float*)carve((size_t)M * D * 4);
  __bf16* h2    = (__bf16*)carve((size_t)M * D * 2);
  __bf16* mlph  = (__bf16*)carve((size_t)M * 4 * D * 2);

  auto blk = [](int n) { return (n + 255) / 256; };

  // weights -> bf16
  cvt_bf16_kernel<<<blk(3 * D * D), 256, 0, stream>>>(qkv_w, wqkv, 3 * D * D);
  cvt_bf16_kernel<<<blk(D * D), 256, 0, stream>>>(proj_w, wproj, D * D);
  cvt_bf16_kernel<<<blk(4 * D * D), 256, 0, stream>>>(mlp_w1, wm1, 4 * D * D);
  cvt_bf16_kernel<<<blk(4 * D * D), 256, 0, stream>>>(mlp_w2, wm2, 4 * D * D);

  // AdaLN conditioning
  silu_kernel<<<blk(B * D), 256, 0, stream>>>(cond, sc, B * D);
  cond_gemm_kernel<<<blk(B * 6 * D), 256, 0, stream>>>(sc, cond_w, cond_b, cbuf);

  // attention branch
  ln_mod_kernel<<<M / 8, 256, 0, stream>>>(x, cbuf, 0 * D, 1 * D, h1);
  gemm_bf16_kernel<<<dim3(M / 64, (3 * D) / 32), 128, 0, stream>>>(
      h1, wqkv, qkv_b, M, 3 * D, D, 0, qkvb, nullptr, nullptr, nullptr);
  transpose_v_kernel<<<blk(M * D), 256, 0, stream>>>(qkvb, vtb);
  attn_kernel<<<dim3(L / 64, 6, B), 128, 0, stream>>>(qkvb, vtb, attnb);
  gemm_bf16_kernel<<<dim3(M / 64, D / 32), 128, 0, stream>>>(
      attnb, wproj, proj_b, M, D, D, 2, nullptr, x2, x, cbuf + 2 * D);

  // MLP branch
  ln_mod_kernel<<<M / 8, 256, 0, stream>>>(x2, cbuf, 3 * D, 4 * D, h2);
  gemm_bf16_kernel<<<dim3(M / 64, (4 * D) / 32), 128, 0, stream>>>(
      h2, wm1, mlp_b1, M, 4 * D, D, 1, mlph, nullptr, nullptr, nullptr);
  gemm_bf16_kernel<<<dim3(M / 64, D / 32), 128, 0, stream>>>(
      mlph, wm2, mlp_b2, M, D, 4 * D, 2, nullptr, out, x2, cbuf + 5 * D);
}
